// RNNLM_57569741635990
// MI455X (gfx1250) — matrix operand order repaired and checked by reading, not compiled, not run on hardware
//
#include <hip/hip_runtime.h>
#include <hip/hip_bf16.h>
#include <stdint.h>

// ---------------------------------------------------------------------------
// Types for WMMA fragments (wave32, 16x16x32 bf16 -> f32) and TDM descriptors
// ---------------------------------------------------------------------------
typedef __attribute__((ext_vector_type(16))) __bf16 bf16x16;
typedef __attribute__((ext_vector_type(8)))  __bf16 bf16x8;
typedef __attribute__((ext_vector_type(8)))  float  f32x8;
typedef __attribute__((ext_vector_type(4)))  unsigned int u32x4;
typedef __attribute__((ext_vector_type(8)))  unsigned int u32x8;

union FragBF16 { bf16x16 v; bf16x8 h[2]; };

// Flat LDS pointers carry the LDS byte offset in addr[31:0] (ISA 10.2).
static __device__ __forceinline__ unsigned lds_off(const void* p) {
  return (unsigned)(uintptr_t)p;
}

// CDNA5 async copy global -> LDS (ASYNCcnt), 16 bytes per lane.
static __device__ __forceinline__ void async_copy_16B(unsigned lds_byte_off,
                                                      const void* gptr) {
  asm volatile("global_load_async_to_lds_b128 %0, %1, off"
               :: "v"(lds_byte_off), "v"((unsigned long long)(uintptr_t)gptr)
               : "memory");
}
static __device__ __forceinline__ void wait_async0() {
  asm volatile("s_wait_asynccnt 0x0" ::: "memory");
}

// ---------------------------------------------------------------------------
// Tensor Data Mover: 2-D tile load, bf16 elements.
//   tile = tile_d1 rows x 32 halves (64 B), row stride in memory = k halves.
//   LDS padding by TDM: pad_interval = 16 DWORDs (one row), pad_amount =
//   4 DWORDs -> LDS row stride 80 B == the [.][40] padded layout used below.
// D# per CDNA5 ISA; 2 SGPR groups (VADDR2/3 NULL, 2-D).
// ---------------------------------------------------------------------------
static __device__ __forceinline__ void tdm_load_tile_2d(unsigned lds_byte_off,
                                                        const void* gptr,
                                                        unsigned k_halves,
                                                        unsigned tile_d1) {
  unsigned long long ga = (unsigned long long)(uintptr_t)gptr;
  u32x4 g0;
  g0[0] = 1u;                                     // count=1 (valid user D#)
  g0[1] = lds_byte_off;                           // lds_addr
  g0[2] = (unsigned)ga;                           // global_addr[31:0]
  g0[3] = ((unsigned)(ga >> 32) & 0x01FFFFFFu)    // global_addr[56:32]
          | 0x80000000u;                          // type=2 ("image")
  u32x8 g1;
  g1[0] = (1u << 16)                              // data_size = 2 bytes
        | (1u << 20)                              // pad_enable
        | (3u << 22)                              // pad_interval: 16 DWORDs
        | (3u << 25);                             // pad_amount:   4 DWORDs
  g1[1] = 0u;                                     // no atomic barrier; dim0 lo16=0
  g1[2] = 0x10u;                                  // tensor_dim0 = 1<<20 (no clip)
  g1[3] = 0x10u | (32u << 16);                    // tensor_dim1 = 1<<20; tile_dim0=32
  g1[4] = tile_d1;                                // tile_dim1 (rows); tile_dim2=0
  g1[5] = k_halves;                               // tensor_dim0_stride[31:0]
  g1[6] = 0u;                                     // stride hi / dim1_stride lo
  g1[7] = 0u;
  asm volatile("tensor_load_to_lds %0, %1" :: "s"(g0), "s"(g1) : "memory");
}

// ---------------------------------------------------------------------------
// fp32 -> bf16 bulk convert (one-time; makes emb L2-resident at 65 MB)
// ---------------------------------------------------------------------------
__global__ __launch_bounds__(256) void cvt_f32_to_bf16(const float* __restrict__ src,
                                                       __bf16* __restrict__ dst,
                                                       long long n) {
  long long i = (long long)blockIdx.x * blockDim.x + threadIdx.x;
  long long step = (long long)gridDim.x * blockDim.x;
  for (; i < n; i += step) dst[i] = (__bf16)src[i];
}

// ---------------------------------------------------------------------------
// Embedding gather: xe[b*T+t, :] = bf16(emb[x[b,t], :])
// ---------------------------------------------------------------------------
__global__ __launch_bounds__(256) void gather_embed(const int* __restrict__ x,
                                                    const float* __restrict__ emb,
                                                    __bf16* __restrict__ xe) {
  const int row = blockIdx.x;               // b*T + t (2048 rows)
  const int idx = x[row];
  const float* s = emb + (size_t)idx * 1024;
  __bf16* d = xe + (size_t)row * 1024;
  for (int c = threadIdx.x; c < 1024; c += 256) d[c] = (__bf16)s[c];
}

// ---------------------------------------------------------------------------
// C[M,N] = A[M,K] @ B[N,K]^T  (bf16 in, f32 out), optional bias0+bias1 per col.
// Block tile 128x128, K-step 32, 8 waves (2Mx4N), wave tile 64x32.
// K-tiles staged to LDS by the Tensor Data Mover (TENSORcnt), double-buffered:
// wave 0 issues the next tile's 2 descriptor loads, waits tensorcnt<=2 so the
// current tile is complete (per-wave in-order TDM) while the next is in flight.
// M,N,K assumed multiples of 128/128/32 (true for both call sites).
// ---------------------------------------------------------------------------
#define BM  128
#define BN  128
#define BK  32
#define BKP 40   // padded row stride (halves): 80B rows, written by TDM padding

__global__ __launch_bounds__(256) void gemm_bf16_nt(const __bf16* __restrict__ A,
                                                    const __bf16* __restrict__ B,
                                                    float* __restrict__ C,
                                                    int M, int N, int K,
                                                    const float* __restrict__ bias0,
                                                    const float* __restrict__ bias1) {
  __shared__ __bf16 sA[2][BM][BKP];
  __shared__ __bf16 sB[2][BN][BKP];

  const int tid  = threadIdx.x;
  const int lane = tid & 31;
  const int wave = tid >> 5;
  const int wm   = wave & 1;        // 2 M-waves
  const int wn   = wave >> 1;       // 4 N-waves
  const int bm   = blockIdx.y * BM;
  const int bn   = blockIdx.x * BN;

  f32x8 acc[4][2];
  #pragma unroll
  for (int i = 0; i < 4; ++i)
    #pragma unroll
    for (int j = 0; j < 2; ++j) acc[i][j] = 0.f;

  const int mlo = lane & 15;
  const int khf = (lane >> 4) * 8;   // A frag: halves {khf..+7, khf+16..+23}
  const int kcl = (lane >> 4) * 16;  // unused; B frag reads the same khf halves as A

  const int nk = K / BK;
  const __bf16* Abase = A + (size_t)bm * K;
  const __bf16* Bbase = B + (size_t)bn * K;

  if (wave == 0) {
    tdm_load_tile_2d(lds_off(&sA[0][0][0]), Abase, (unsigned)K, BM);
    tdm_load_tile_2d(lds_off(&sB[0][0][0]), Bbase, (unsigned)K, BN);
  }

  int cur = 0;
  for (int ki = 0; ki < nk; ++ki) {
    __syncthreads();   // buffer cur^1 fully consumed before TDM overwrites it
    if (wave == 0) {
      if (ki + 1 < nk) {
        tdm_load_tile_2d(lds_off(&sA[cur ^ 1][0][0]),
                         Abase + (size_t)(ki + 1) * BK, (unsigned)K, BM);
        tdm_load_tile_2d(lds_off(&sB[cur ^ 1][0][0]),
                         Bbase + (size_t)(ki + 1) * BK, (unsigned)K, BN);
        __builtin_amdgcn_s_wait_tensorcnt(2);   // tile ki landed; ki+1 in flight
      } else {
        __builtin_amdgcn_s_wait_tensorcnt(0);
      }
    } else if (ki + 2 < nk) {
      // Warm L2 for tile ki+2 while wave 0 talks to the TDM.
      int r = (wave - 1) * 32 + lane;           // 0..223
      const __bf16* p = (r < BM) ? Abase + (size_t)r * K + (ki + 2) * BK
                                 : Bbase + (size_t)(r - BM) * K + (ki + 2) * BK;
      __builtin_prefetch(p, 0, 1);
    }
    __syncthreads();   // TDM data visible to all waves

    const __bf16 (*pA)[BKP] = sA[cur];
    const __bf16 (*pB)[BKP] = sB[cur];

    FragBF16 a[4], b[2];
    #pragma unroll
    for (int i = 0; i < 4; ++i) {
      int r = wm * 64 + i * 16 + mlo;
      a[i].h[0] = *(const bf16x8*)&pA[r][khf];
      a[i].h[1] = *(const bf16x8*)&pA[r][khf + 16];
    }
    #pragma unroll
    for (int j = 0; j < 2; ++j) {
      int r = wn * 32 + j * 16 + mlo;
      b[j].h[0] = *(const bf16x8*)&pB[r][khf];
      b[j].h[1] = *(const bf16x8*)&pB[r][khf + 16];
    }
    #pragma unroll
    for (int i = 0; i < 4; ++i)
      #pragma unroll
      for (int j = 0; j < 2; ++j)
        acc[i][j] = __builtin_amdgcn_wmma_f32_16x16x32_bf16(
            false, a[i].v, false, b[j].v, (short)0, acc[i][j], false, false);
    cur ^= 1;
  }

  // Epilogue: C/D layout -> VGPR e holds rows {e, e+8}, lane&15 = column.
  const int mhi = (lane >> 4) * 8;
  const int nlo = lane & 15;
  #pragma unroll
  for (int i = 0; i < 4; ++i) {
    #pragma unroll
    for (int j = 0; j < 2; ++j) {
      int col = bn + wn * 32 + j * 16 + nlo;
      float badd = bias0 ? (bias0[col] + bias1[col]) : 0.f;
      #pragma unroll
      for (int e = 0; e < 8; ++e) {
        int row = bm + wm * 64 + i * 16 + mhi + e;
        C[(size_t)row * N + col] = acc[i][j][e] + badd;
      }
    }
  }
}

// ---------------------------------------------------------------------------
// Persistent LSTM scan. 64 WGs x 128 threads (4 waves). WG `wg` owns hidden
// units [wg*16, wg*16+16); wave g computes gate type g (i,f,g,o) as a 16x16
// WMMA tile over K=1024. h ping-pongs in global (L2-resident) and is staged
// into LDS via async-to-LDS (ASYNCcnt). One device barrier per timestep.
// Batch 8 padded to M=16 (pad rows stay zero).
// ---------------------------------------------------------------------------
__global__ __launch_bounds__(128) void lstm_scan(const float* __restrict__ xg,   // [2048,4096]
                                                 const __bf16* __restrict__ whh, // [4096,1024]
                                                 __bf16* __restrict__ hs,        // [2048,1024]
                                                 __bf16* __restrict__ hbuf,      // [2][16][1024]
                                                 unsigned* __restrict__ bar) {   // [256]
  __shared__ __bf16 sH[16][1032];   // padded rows: 2064B stride, conflict-free
  __shared__ float  sG[4][16][16];  // gate tiles (i,f,g,o)
  __shared__ float  sC[8][16];      // cell state for this WG's hidden slice

  const int tid  = threadIdx.x;
  const int lane = tid & 31;
  const int gate = tid >> 5;                 // wave id == gate type
  const int wg   = blockIdx.x;               // hidden slice base wg*16
  const int nlo  = lane & 15;
  const int mhi  = (lane >> 4) * 8;
  const int khf  = (lane >> 4) * 8;
  const int kcl  = (lane >> 4) * 16;

  sC[tid >> 4][tid & 15] = 0.f;

  const int gcol = gate * 1024 + wg * 16 + nlo;      // column in the 4H gate space
  const __bf16* brow = whh + (size_t)gcol * 1024;    // W_hh row = B column (K-contig)

  for (int t = 0; t < 256; ++t) {
    __syncthreads();
    // Stage h_t (16x1024 bf16 = 32 KB) into LDS with async-to-LDS copies.
    const __bf16* hc = hbuf + (size_t)(t & 1) * (16 * 1024);
    #pragma unroll
    for (int u = 0; u < 16; ++u) {
      int c = tid + u * 128;                  // 2048 x 16B chunks
      int r = c >> 7, col = (c & 127) * 8;
      async_copy_16B(lds_off(&sH[r][col]), hc + r * 1024 + col);
    }
    wait_async0();
    __syncthreads();

    // acc starts from precomputed x_gates (+biases); pad rows (8..15) = 0.
    f32x8 acc;
    if (lane < 16) {
      #pragma unroll
      for (int e = 0; e < 8; ++e)
        acc[e] = xg[((size_t)e * 256 + t) * 4096 + gcol];
    } else {
      #pragma unroll
      for (int e = 0; e < 8; ++e) acc[e] = 0.f;
    }

    // gates += h_t @ W_hh^T : 32 chained WMMAs (K=1024).
    #pragma unroll 4
    for (int kk = 0; kk < 32; ++kk) {
      FragBF16 a, b;
      a.h[0] = *(const bf16x8*)&sH[nlo][kk * 32 + khf];
      a.h[1] = *(const bf16x8*)&sH[nlo][kk * 32 + khf + 16];
      const __bf16* bp = brow + kk * 32 + khf;       // L2-resident W_hh
      b.h[0] = *(const bf16x8*)bp;
      b.h[1] = *(const bf16x8*)(bp + 16);
      acc = __builtin_amdgcn_wmma_f32_16x16x32_bf16(false, a.v, false, b.v,
                                                    (short)0, acc, false, false);
    }

    #pragma unroll
    for (int e = 0; e < 8; ++e) sG[gate][mhi + e][nlo] = acc[e];
    __syncthreads();

    // LSTM cell update for (m = tid/16 in 0..7, j = tid%16).
    {
      const int m = tid >> 4, j = tid & 15;
      float iv = 1.f / (1.f + __expf(-sG[0][m][j]));
      float fv = 1.f / (1.f + __expf(-sG[1][m][j]));
      float gv = tanhf(sG[2][m][j]);
      float ov = 1.f / (1.f + __expf(-sG[3][m][j]));
      float cv = fv * sC[m][j] + iv * gv;
      sC[m][j] = cv;
      __bf16 hb = (__bf16)(ov * tanhf(cv));
      __bf16* hn = hbuf + (size_t)((t + 1) & 1) * (16 * 1024);
      hn[m * 1024 + wg * 16 + j] = hb;                          // next-step h
      hs[((size_t)m * 256 + t) * 1024 + wg * 16 + j] = hb;      // logits A matrix
    }

    // Device-wide barrier: fresh counter slot per step (memset to 0 each launch).
    __threadfence();
    __syncthreads();
    if (tid == 0)
      __hip_atomic_fetch_add(&bar[t], 1u, __ATOMIC_RELEASE, __HIP_MEMORY_SCOPE_AGENT);
    while (__hip_atomic_load(&bar[t], __ATOMIC_ACQUIRE, __HIP_MEMORY_SCOPE_AGENT) < 64u)
      __builtin_amdgcn_s_sleep(2);
  }
}

// ---------------------------------------------------------------------------
// Launch: cvt -> gather -> x_gates GEMM -> persistent scan -> logits GEMM
// ---------------------------------------------------------------------------
extern "C" void kernel_launch(void* const* d_in, const int* in_sizes, int n_in,
                              void* d_out, int out_size, void* d_ws, size_t ws_size,
                              hipStream_t stream) {
  const int*   x   = (const int*)  d_in[0];
  const float* emb = (const float*)d_in[1];
  const float* Wih = (const float*)d_in[2];
  const float* Whh = (const float*)d_in[3];
  const float* bih = (const float*)d_in[4];
  const float* bhh = (const float*)d_in[5];
  float* out = (float*)d_out;
  (void)in_sizes; (void)n_in; (void)out_size; (void)ws_size;

  char* w = (char*)d_ws;
  size_t o = 0;
  auto carve = [&](size_t bytes) -> char* {
    char* p = w + o; o += (bytes + 255) & ~(size_t)255; return p;
  };
  __bf16*  emb_bf = (__bf16*)carve((size_t)32000 * 1024 * 2); // 65.5 MB
  __bf16*  wih_bf = (__bf16*)carve((size_t)4096  * 1024 * 2); //  8.4 MB
  __bf16*  whh_bf = (__bf16*)carve((size_t)4096  * 1024 * 2); //  8.4 MB
  __bf16*  xe_bf  = (__bf16*)carve((size_t)2048  * 1024 * 2); //  4.2 MB
  float*   xgw    = (float*) carve((size_t)2048  * 4096 * 4); // 33.6 MB
  __bf16*  hs_bf  = (__bf16*)carve((size_t)2048  * 1024 * 2); //  4.2 MB
  __bf16*  hbuf   = (__bf16*)carve((size_t)2 * 16 * 1024 * 2);
  unsigned* bar   = (unsigned*)carve((size_t)256 * 4);

  cvt_f32_to_bf16<<<8192, 256, 0, stream>>>(emb, emb_bf, (long long)32000 * 1024);
  cvt_f32_to_bf16<<<2048, 256, 0, stream>>>(Wih, wih_bf, (long long)4096 * 1024);
  cvt_f32_to_bf16<<<2048, 256, 0, stream>>>(Whh, whh_bf, (long long)4096 * 1024);
  gather_embed<<<2048, 256, 0, stream>>>(x, emb, xe_bf);

  // x_gates = xe @ W_ih^T + (b_ih + b_hh)   [2048 x 4096]
  gemm_bf16_nt<<<dim3(4096 / BN, 2048 / BM), 256, 0, stream>>>(
      xe_bf, wih_bf, xgw, 2048, 4096, 1024, bih, bhh);

  hipMemsetAsync(hbuf, 0, (size_t)2 * 16 * 1024 * 2, stream);
  hipMemsetAsync(bar,  0, (size_t)256 * 4, stream);
  lstm_scan<<<64, 128, 0, stream>>>(xgw, whh_bf, hs_bf, hbuf, bar);

  // logits = hs @ emb^T   [2048 x 32000]
  gemm_bf16_nt<<<dim3(32000 / BN, 2048 / BM), 256, 0, stream>>>(
      hs_bf, emb_bf, out, 2048, 32000, 1024, nullptr, nullptr);
}
